// Brain3DNetwork_19928648253664
// MI455X (gfx1250) — matrix operand order repaired and checked by reading, not compiled, not run on hardware
//
#include <hip/hip_runtime.h>
#include <stdint.h>

// ---------------------------------------------------------------------------
// Problem constants (match the reference: L=32, H=128, W=128, R=1, 10 steps)
// ---------------------------------------------------------------------------
#define L_ 32
#define H_ 128
#define W_ 128
#define N_ (L_ * H_ * W_)     // 524288 neurons
#define NOFF 26               // 26-neighbor stencil
#define NSTEPS 10

typedef __attribute__((ext_vector_type(16))) _Float16 v16h;
typedef __attribute__((ext_vector_type(8)))  float    v8f;

// Offset tables in the exact loop order of the reference edge builder:
// dl in -1..1, dh in -1..1, dw in -1..1, skipping (0,0,0).
constexpr int kDL[NOFF] = {-1,-1,-1,-1,-1,-1,-1,-1,-1,  0,0,0,0,0,0,0,0,  1,1,1,1,1,1,1,1,1};
constexpr int kDH[NOFF] = {-1,-1,-1, 0, 0, 0, 1, 1, 1, -1,-1,-1,0,0,1,1,1, -1,-1,-1,0,0,0,1,1,1};
constexpr int kDW[NOFF] = {-1, 0, 1,-1, 0, 1,-1, 0, 1, -1, 0, 1,-1,1,-1,0,1, -1,0,1,-1,0,1,-1,0,1};

struct EdgeTables { long long base[NOFF]; };
constexpr EdgeTables makeTables() {
    EdgeTables t{};
    long long b = 0;
    for (int k = 0; k < NOFF; ++k) {
        t.base[k] = b;
        int adl = kDL[k] < 0 ? -kDL[k] : kDL[k];
        int adh = kDH[k] < 0 ? -kDH[k] : kDH[k];
        int adw = kDW[k] < 0 ? -kDW[k] : kDW[k];
        b += (long long)(L_ - adl) * (H_ - adh) * (W_ - adw);
    }
    return t;
}
constexpr EdgeTables kET = makeTables();

// ---------------------------------------------------------------------------
// Kernel 1: repack per-edge exponents into fp16 weights stored in WMMA
// A-fragment order. wtab16[tile][lane][j], 32 lanes * 16 halves = 1 KB/tile.
// A-fragment layout for v_wmma_f32_16x16x32_f16 (16-bit A, 16x32):
//   lane = (lanehalf<<4) | m,  m = dst row (M),  lanehalf = (k>>3)&1
//   j    = (k & 7) + (k >= 16 ? 8 : 0)
// Missing/out-of-bounds edges and padding (k=26..31) store weight 0.
// Also computes baseline[dst] = 1e-8 * sum(w) over existing incoming edges
// (the reference's +1e-8 background term, split out exactly), and zeroes the
// per-step spike counters.
// ---------------------------------------------------------------------------
__global__ void repack_weights(const int* __restrict__ vexp,
                               _Float16* __restrict__ wtab16,
                               float* __restrict__ baseline,
                               int* __restrict__ counters) {
    int dst = blockIdx.x * blockDim.x + threadIdx.x;
    if (dst >= N_) return;
    if (dst < 16) counters[dst] = 0;

    const int w = dst & (W_ - 1);
    const int h = (dst >> 7) & (H_ - 1);
    const int l = dst >> 14;
    const int tile = dst >> 4;
    const int m = dst & 15;
    _Float16* frag = wtab16 + (size_t)tile * 512;

    float sumw = 0.0f;
#pragma unroll
    for (int k = 0; k < NOFF; ++k) {
        const int dl = kDL[k], dh = kDH[k], dw = kDW[k];
        const int sl = l - dl, sh = h - dh, sw = w - dw;
        float wv = 0.0f;
        if ((unsigned)sl < (unsigned)L_ && (unsigned)sh < (unsigned)H_ &&
            (unsigned)sw < (unsigned)W_) {
            const int l0 = (dl < 0) ? -dl : 0;
            const int h0 = (dh < 0) ? -dh : 0;
            const int w0 = (dw < 0) ? -dw : 0;
            const int nh = H_ - (dh < 0 ? -dh : dh);
            const int nw = W_ - (dw < 0 ? -dw : dw);
            const long long idx =
                ((long long)(sl - l0) * nh + (sh - h0)) * nw + (sw - w0);
            const int e = vexp[kET.base[k] + idx];
            wv = (float)(1 << e);                 // 2^exp, exact in fp16
            sumw += wv;
        }
        const int lanehalf = (k >> 3) & 1;
        const int j = (k & 7) + (k >= 16 ? 8 : 0);
        frag[(size_t)(((lanehalf << 4) | m) * 16 + j)] = (_Float16)wv;
    }
    // padding slots k = 26..31 live at lanehalf=1, j=10..15
#pragma unroll
    for (int j = 10; j < 16; ++j)
        frag[(size_t)((16 | m) * 16 + j)] = (_Float16)0.0f;

    baseline[dst] = 1e-8f * sumw;
}

// ---------------------------------------------------------------------------
// Async global -> LDS copy of one float (CDNA5 GLOBAL_LOAD_ASYNC_TO_LDS_B32,
// GVS addressing: SGPR64 base + VGPR32 byte offset). Tracked by ASYNCcnt.
// ---------------------------------------------------------------------------
__device__ __forceinline__ void async_copy_f32(unsigned lds_byte_addr,
                                               unsigned gbl_byte_off,
                                               const float* base) {
    asm volatile("global_load_async_to_lds_b32 %0, %1, %2"
                 :
                 : "v"(lds_byte_addr), "v"(gbl_byte_off),
                   "s"((unsigned long long)(uintptr_t)base)
                 : "memory");
}
__device__ __forceinline__ void wait_async_zero() {
    asm volatile("s_wait_asynccnt 0" ::: "memory");
}

// ---------------------------------------------------------------------------
// Kernel 2: one LIF time-step. One 256-thread block (8 wave32) per (l,h) row
// of 128 dst neurons; one wave per 16-dst tile.
//   Stage:  9 halo rows (l+dl, h+dh) x 128 prev-spike floats are copied
//           global->LDS with async-to-LDS ops; each LDS row is padded with a
//           zero column on both sides so w-boundary gathers need no checks.
//   A (16x32 fp16): weights, one 32 B fragment-ready vector load per lane.
//   B (32x16 fp16): column n = dst n's 26 neighbor spikes, unconditional
//           ds_load gathers from the staged halo.
//   D = A x B (f32 accum): diagonal D[i][i] = sum_k w[i][k]*spike[i][k].
// Diagonal extracted via LDS; fp16-rounded (total+baseline) feeds the LIF.
// counters[step] collects spike population for the next step's zero-skip.
// ---------------------------------------------------------------------------
__global__ void lif_step(const float* __restrict__ ext,
                         const _Float16* __restrict__ wtab16,
                         const float* __restrict__ baseline,
                         int* __restrict__ counters,
                         float* __restrict__ spikes_hist,
                         float* __restrict__ volt_hist,
                         int step) {
    const float decay = 0.96721610048f;   // exp(-1/30)
    const float gain  = 0.98351698554f;   // 30*(1-exp(-1/30))

    __shared__ float lds_prev[9 * 130];   // 9 halo rows, cols: [0]=0,1..128=data,[129]=0
    __shared__ float dbuf[8][32][8];      // [wave][lane][vgpr] D staging

    const int tid = threadIdx.x;
    const int lane = tid & 31;
    const int waveInBlk = tid >> 5;
    const int row = blockIdx.x;                  // (l,h) row id, 0..4095
    const int l = row >> 7;
    const int h = row & (H_ - 1);
    const int tile = row * 8 + waveInBlk;

    const int prevcnt = (step == 0) ? 0 : counters[step - 1];   // grid-uniform
    const float* __restrict__ prev = spikes_hist + (size_t)(step - 1) * N_;

    if (prevcnt > 0) {
        // Zero the pad columns of all 9 rows.
        if (tid < 18) {
            const int r = tid >> 1;
            lds_prev[r * 130 + ((tid & 1) ? 129 : 0)] = 0.0f;
        }
        // Stage 9 x 128 floats: valid rows via async-to-LDS, OOB rows zeroed.
#pragma unroll
        for (int p = 0; p < 5; ++p) {
            const int c = tid + p * 256;
            if (c < 9 * 128) {
                const int r = c >> 7;            // halo row 0..8
                const int x = c & 127;           // w position
                const int sl = l + (r / 3) - 1;
                const int sh = h + (r % 3) - 1;
                float* dstp = &lds_prev[r * 130 + 1 + x];
                if ((unsigned)sl < (unsigned)L_ && (unsigned)sh < (unsigned)H_) {
                    const unsigned goff = (unsigned)(((sl << 14) + (sh << 7) + x) * 4);
                    async_copy_f32((unsigned)(uintptr_t)dstp, goff, prev);
                } else {
                    *dstp = 0.0f;
                }
            }
        }
        wait_async_zero();
    }
    __syncthreads();

    if (prevcnt > 0) {
        // ---- A fragment: 16 fp16 weights per lane, fragment-ready.
        v16h a = *reinterpret_cast<const v16h*>(wtab16 + (size_t)tile * 512 +
                                                (size_t)lane * 16);
        // Stream the next row's weights toward the caches (global_prefetch_b8).
        int pf = tile + 8; if (pf >= N_ / 16) pf = tile;
        __builtin_prefetch(wtab16 + (size_t)pf * 512, 0, 0);

        // ---- B fragment: unconditional gathers from the padded LDS halo.
        const int n = lane & 15;
        const int half = lane >> 4;
        const int wcol = waveInBlk * 16 + n;     // dst w within the row
        v16h b;
#pragma unroll
        for (int j = 0; j < 16; ++j) {
            const int k = (j < 8) ? (half * 8 + j) : (16 + half * 8 + (j - 8));
            float v = 0.0f;
            if (k < NOFF) {
                const int ridx = (kDL[k] + 1) * 3 + (kDH[k] + 1);
                v = lds_prev[ridx * 130 + 1 + wcol - kDW[k]];
            }
            b[j] = (_Float16)v;
        }

        // ---- Matrix op: D = A x B + 0, f32 accumulation (exact here).
        v8f c = {};
        c = __builtin_amdgcn_wmma_f32_16x16x32_f16(
                /*neg_a=*/false, a, /*neg_b=*/false, b,
                /*c_mod=*/(short)0, c, /*reuse_a=*/false, /*reuse_b=*/false);
#pragma unroll
        for (int r = 0; r < 8; ++r) dbuf[waveInBlk][lane][r] = c[r];
    }
    __syncthreads();

    // ---- Diagonal D[m][m]: lane = m (+16 if m>=8), vgpr = m&7 (branchless).
    const int m = lane;
    const bool active = (lane < 16);
    float spike = 0.0f;
    if (active) {
        float tot = 0.0f;
        const int dst = tile * 16 + m;
        if (prevcnt > 0) {
            const float diag = dbuf[waveInBlk][m + ((m & 8) << 1)][m & 7];
            tot = diag + baseline[dst];
            tot = (float)(_Float16)tot;          // reference's fp16 cast
        }
        const float vprev =
            (step == 0) ? 0.0f : volt_hist[(size_t)(step - 1) * N_ + dst];
        const float inp = tot + ext[dst];
        float v = vprev * decay + inp * gain;
        spike = (v >= 0.1f) ? 1.0f : 0.0f;
        v = (spike != 0.0f) ? 0.0f : v;
        spikes_hist[(size_t)step * N_ + dst] = spike;
        volt_hist[(size_t)step * N_ + dst] = v;
    }
    const unsigned long long bal = __ballot(active && (spike != 0.0f));
    if (lane == 0) atomicAdd(&counters[step], __popcll(bal));
}

// ---------------------------------------------------------------------------
// kernel_launch
//   d_in: [0] external_input f32[N], [1] src_ids i32[E], [2] dst_ids i32[E],
//         [3] values_exp i32[E], [4] num_steps (scalar, fixed = 10)
//   d_out: spikes_hist f32[10*N] | volt_hist f32[10*N] | values_exp (raw)
//   d_ws : [0,64)            counters
//          [64, +4N)         baseline f32
//          [64+4N, +64N)     wtab16 fp16 (~34 MB total)
// ---------------------------------------------------------------------------
extern "C" void kernel_launch(void* const* d_in, const int* in_sizes, int n_in,
                              void* d_out, int out_size, void* d_ws, size_t ws_size,
                              hipStream_t stream) {
    const float* ext  = (const float*)d_in[0];
    const int*   vexp = (const int*)d_in[3];
    const int    E    = in_sizes[3];

    float* spikes_hist = (float*)d_out;
    float* volt_hist   = spikes_hist + (size_t)NSTEPS * N_;

    // Third output: values_exp passed through unchanged (raw 4-byte copy).
    hipMemcpyAsync((char*)d_out + (size_t)2 * NSTEPS * N_ * sizeof(float), vexp,
                   (size_t)E * sizeof(int), hipMemcpyDeviceToDevice, stream);

    char* ws = (char*)d_ws;
    int*      counters = (int*)ws;
    float*    baseline = (float*)(ws + 64);
    _Float16* wtab16   = (_Float16*)(ws + 64 + (size_t)N_ * sizeof(float));

    repack_weights<<<N_ / 256, 256, 0, stream>>>(vexp, wtab16, baseline, counters);

    const int nBlocks = N_ / 128;         // one block per (l,h) row
    for (int s = 0; s < NSTEPS; ++s) {
        lif_step<<<nBlocks, 256, 0, stream>>>(ext, wtab16, baseline, counters,
                                              spikes_hist, volt_hist, s);
    }
}
